// BigramLanguageModel_81449759802294
// MI455X (gfx1250) — matrix operand order repaired and checked by reading, not compiled, not run on hardware
//
#include <hip/hip_runtime.h>
#include <hip/hip_bf16.h>

// ---------------------------------------------------------------------------
// GPT forward (B=4,T=2048,E=1024,H=16,Dh=64,L=4,FF=4096,V=32000) on gfx1250.
// GEMMs: v_wmma_f32_16x16x32_f16, async-to-LDS double-buffered A staging,
// weights pre-packed in WMMA-fragment-native order (direct global B frags).
// Wave tile 32x(NT*16): NT=4 for large-N GEMMs, NT=2 for the N=64 QKV GEMM.
// ---------------------------------------------------------------------------

typedef __attribute__((ext_vector_type(16))) _Float16 v16h;
typedef __attribute__((ext_vector_type(8)))  _Float16 v8h;
typedef __attribute__((ext_vector_type(8)))  float    v8f;

union FragH { v16h v; v8h h[2]; };

#define BM 128
#define BKK 32
#define LDSS 40   // f16 row pitch; keeps 8-half chunks 16B aligned (80B pitch)

// ---------------------------------------------------------------------------
// CDNA5 async global->LDS copy (16 bytes/lane), tracked by ASYNCcnt.
// ---------------------------------------------------------------------------
__device__ __forceinline__ void async_copy_b128(uint32_t lds_byte_off,
                                                const _Float16* gptr)
{
    unsigned long long ga = (unsigned long long)gptr;
    asm volatile("global_load_async_to_lds_b128 %0, %1, off"
                 :: "v"(lds_byte_off), "v"(ga) : "memory");
}
__device__ __forceinline__ void wait_async_zero()
{
    asm volatile("s_wait_asynccnt 0x0" ::: "memory");
}

// ---------------------------------------------------------------------------
// GEMM: C[M,N] = A[M,K](f16 row-major) x W^T + bias, optional exact GELU.
// Bp = pre-packed weights: [N/16][K/32][lane(32)][16 halfs], lane fragment
// order: n = lane%16, k = hf*16+e (contiguous k, same order as the A fragments).
// Block: 256 thr = 8 waves as 4(M) x 2(N); wave tile 32 x NT*16.
// M%128==0, N%(32*NT)==0, K%32==0 at every call site.
// ---------------------------------------------------------------------------
template<int NT>
__global__ __launch_bounds__(256) void gemm_f16_wmma(
    const _Float16* __restrict__ A, const _Float16* __restrict__ Bp,
    float* __restrict__ Cf, _Float16* __restrict__ Ch,
    const float* __restrict__ bias, int M, int N, int K, int act_gelu)
{
    __shared__ _Float16 As[2 * BM * LDSS];

    const int tid   = threadIdx.x;
    const int lane  = tid & 31;
    const int wave  = tid >> 5;
    const int hf    = lane >> 4;
    const int lm    = lane & 15;
    const int bm    = blockIdx.x, bn = blockIdx.y;
    const int waveM = wave >> 1,  waveN = wave & 1;
    const int rowb  = waveM * 32;
    const int ktiles = K >> 5;
    const int nk     = ktiles;
    const uint32_t As_base = (uint32_t)(uintptr_t)(&As[0]);

    // global 16-wide n-tile index of this wave's first column fragment
    const int gnt0 = (bn * 2 + waveN) * NT;

    v8f zero = {};
    v8f acc[2][NT];
    #pragma unroll
    for (int mt = 0; mt < 2; ++mt)
        #pragma unroll
        for (int nt = 0; nt < NT; ++nt) acc[mt][nt] = zero;

    // ---- preamble: async-stage first A tile into buffer 0 ----
    #pragma unroll
    for (int it = 0; it < 2; ++it) {
        int i  = tid + it * 256;
        int r  = i >> 2;
        int c8 = (i & 3) * 8;
        uint32_t loff = As_base + (uint32_t)((r * LDSS + c8) * 2);
        async_copy_b128(loff, A + (size_t)(bm * BM + r) * K + c8);
    }

    const v16h* Bv = (const v16h*)Bp;

    for (int kt = 0; kt < nk; ++kt) {
        wait_async_zero();      // my async copies (incl. current buffer) done
        __syncthreads();        // everyone's copies done / prior reads done

        if (kt + 1 < nk) {      // overlap: stage next tile into other buffer
            int nb  = (kt + 1) & 1;
            int k0n = (kt + 1) * BKK;
            #pragma unroll
            for (int it = 0; it < 2; ++it) {
                int i  = tid + it * 256;
                int r  = i >> 2;
                int c8 = (i & 3) * 8;
                uint32_t loff = As_base + (uint32_t)(((nb * BM + r) * LDSS + c8) * 2);
                async_copy_b128(loff, A + (size_t)(bm * BM + r) * K + k0n + c8);
            }
            __builtin_prefetch(Bv + ((size_t)gnt0 * ktiles + kt + 1) * 32 + lane, 0, 1);
        }

        const _Float16* Ab = As + (kt & 1) * BM * LDSS;

        // ---- A fragments: two contiguous ds_load_b128 per lane ----
        FragH a[2], b[NT];
        #pragma unroll
        for (int mt = 0; mt < 2; ++mt) {
            int r = rowb + mt * 16 + lm;
            a[mt].h[0] = *(const v8h*)(&Ab[r * LDSS + hf * 16]);
            a[mt].h[1] = *(const v8h*)(&Ab[r * LDSS + hf * 16 + 8]);
        }
        // ---- B fragments: one coalesced 32B global load per lane each ----
        #pragma unroll
        for (int nt = 0; nt < NT; ++nt)
            b[nt].v = Bv[((size_t)(gnt0 + nt) * ktiles + kt) * 32 + lane];

        #pragma unroll
        for (int mt = 0; mt < 2; ++mt)
            #pragma unroll
            for (int nt = 0; nt < NT; ++nt)
                acc[mt][nt] = __builtin_amdgcn_wmma_f32_16x16x32_f16(
                    false, a[mt].v, false, b[nt].v,
                    (short)0, acc[mt][nt], false, false);
    }

    // ---- epilogue: C layout VGPR r -> M = hf*8 + r, N = lane%16 ----
    #pragma unroll
    for (int mt = 0; mt < 2; ++mt) {
        #pragma unroll
        for (int nt = 0; nt < NT; ++nt) {
            int col = (gnt0 + nt) * 16 + lm;
            float bb = bias ? bias[col] : 0.f;
            #pragma unroll
            for (int r = 0; r < 8; ++r) {
                int row = bm * BM + rowb + mt * 16 + hf * 8 + r;
                float vv = acc[mt][nt][r] + bb;
                if (act_gelu) vv = 0.5f * vv * (1.f + erff(vv * 0.70710678118654752f));
                size_t o = (size_t)row * N + col;
                if (Ch) Ch[o] = (_Float16)vv;
                else    Cf[o] = vv;
            }
        }
    }
}

// ---------------------------------------------------------------------------
// Pack W[N,K] f32 (row-major, y = x@W^T) into WMMA-B fragment order f16:
// P[nt][kt][lane][e]  with n = nt*16 + lane%16, k = kt*32 + (lane/16)*16 + e.
// blockIdx.y batches layers.
// ---------------------------------------------------------------------------
__global__ void pack_w_kernel(const float* __restrict__ in,
                              _Float16* __restrict__ out, int N, int K)
{
    const float* ip = in  + (size_t)blockIdx.y * N * K;
    _Float16*    op = out + (size_t)blockIdx.y * N * K;
    int i = blockIdx.x * 256 + threadIdx.x;
    if (i >= N * K) return;
    int ktiles = K >> 5;
    int e    = i & 15;
    int lane = (i >> 4) & 31;
    int tile = i >> 9;
    int kt   = tile % ktiles;
    int nt   = tile / ktiles;
    int n = nt * 16 + (lane & 15);
    int k = kt * 32 + (lane >> 4) * 16 + e;
    op[i] = (_Float16)ip[(size_t)n * K + k];
}

// ---------------------------------------------------------------------------
// Flash-style causal attention: one wave per query row, online softmax.
// q,k,v,o layout: [B,T,H,Dh] f32, Dh=64, scale=1/8.
// ---------------------------------------------------------------------------
__global__ __launch_bounds__(256) void attention_kernel(
    const float* __restrict__ q, const float* __restrict__ k,
    const float* __restrict__ v, float* __restrict__ o, int T, int H)
{
    __shared__ float sq[8][64];
    const int tid = threadIdx.x, lane = tid & 31, wave = tid >> 5;
    const int t = blockIdx.x * 8 + wave;
    const int b = blockIdx.y / H, h = blockIdx.y % H;
    const size_t rowbase = ((size_t)(b * T + t) * H + h) * 64;

    sq[wave][2 * lane]     = q[rowbase + 2 * lane];
    sq[wave][2 * lane + 1] = q[rowbase + 2 * lane + 1];
    __syncthreads();

    const int d0 = 2 * lane;
    float m = -__builtin_inff(), l = 0.f, acc0 = 0.f, acc1 = 0.f;

    for (int j0 = 0; j0 <= t; j0 += 32) {
        int j = j0 + lane;
        float s = -__builtin_inff();
        if (j <= t) {
            const float4* kp = (const float4*)(k + ((size_t)(b * T + j) * H + h) * 64);
            float dot = 0.f;
            #pragma unroll
            for (int d = 0; d < 16; ++d) {
                float4 kv = kp[d];
                dot += kv.x * sq[wave][4 * d]     + kv.y * sq[wave][4 * d + 1]
                     + kv.z * sq[wave][4 * d + 2] + kv.w * sq[wave][4 * d + 3];
            }
            s = dot * 0.125f;
        }
        float cm = s;
        #pragma unroll
        for (int off = 16; off > 0; off >>= 1) cm = fmaxf(cm, __shfl_xor(cm, off));
        float mnew = fmaxf(m, cm);
        float corr = expf(m - mnew);
        float p    = expf(s - mnew);
        float psum = p;
        #pragma unroll
        for (int off = 16; off > 0; off >>= 1) psum += __shfl_xor(psum, off);
        l = l * corr + psum;
        acc0 *= corr; acc1 *= corr;
        for (int jj = 0; jj < 32; ++jj) {
            float pv = __shfl(p, jj);
            const float* vp = v + ((size_t)(b * T + j0 + jj) * H + h) * 64;
            acc0 += pv * vp[d0];
            acc1 += pv * vp[d0 + 1];
        }
        m = mnew;
    }
    float inv = 1.f / l;
    o[rowbase + d0]     = acc0 * inv;
    o[rowbase + d0 + 1] = acc1 * inv;
}

// ---------------------------------------------------------------------------
// LayerNorm over E=1024 per row: out = LN(inp (+resid)) * g + b
// ---------------------------------------------------------------------------
__global__ __launch_bounds__(256) void ln_kernel(
    const float* __restrict__ inp, const float* __restrict__ resid,
    const float* __restrict__ g, const float* __restrict__ bta,
    float* __restrict__ out)
{
    __shared__ float red[256];
    const int row = blockIdx.x, tid = threadIdx.x;
    const size_t base = (size_t)row * 1024;
    float v[4];
    #pragma unroll
    for (int i = 0; i < 4; ++i) {
        int e = tid + i * 256;
        v[i] = inp[base + e] + (resid ? resid[base + e] : 0.f);
    }
    float s = v[0] + v[1] + v[2] + v[3];
    red[tid] = s; __syncthreads();
    for (int off = 128; off > 0; off >>= 1) {
        if (tid < off) red[tid] += red[tid + off];
        __syncthreads();
    }
    float mean = red[0] * (1.f / 1024.f);
    __syncthreads();
    float s2 = 0.f;
    #pragma unroll
    for (int i = 0; i < 4; ++i) { float d = v[i] - mean; s2 += d * d; }
    red[tid] = s2; __syncthreads();
    for (int off = 128; off > 0; off >>= 1) {
        if (tid < off) red[tid] += red[tid + off];
        __syncthreads();
    }
    float rstd = rsqrtf(red[0] * (1.f / 1024.f) + 1e-5f);
    #pragma unroll
    for (int i = 0; i < 4; ++i) {
        int e = tid + i * 256;
        out[base + e] = (v[i] - mean) * rstd * g[e] + bta[e];
    }
}

// ---------------------------------------------------------------------------
__global__ void embed_kernel(const int* __restrict__ idx,
                             const float* __restrict__ tok,
                             const float* __restrict__ pos,
                             float* __restrict__ x, int T)
{
    int row = blockIdx.x;
    int t = row % T;
    int ix = idx[row];
    size_t tb = (size_t)ix * 1024, pb = (size_t)t * 1024, xb = (size_t)row * 1024;
    for (int e = threadIdx.x; e < 1024; e += 256)
        x[xb + e] = tok[tb + e] + pos[pb + e];
}

__global__ void cvt_f16_kernel(const float* __restrict__ in,
                               _Float16* __restrict__ out, int n)
{
    int i = blockIdx.x * 256 + threadIdx.x;
    if (i < n) out[i] = (_Float16)in[i];
}

// ---------------------------------------------------------------------------
// Deterministic log-softmax NLL loss (two-stage).
// ---------------------------------------------------------------------------
__global__ __launch_bounds__(256) void loss_row_kernel(
    const float* __restrict__ logits, const int* __restrict__ targets,
    float* __restrict__ rowloss, int V)
{
    __shared__ float red[256];
    const int row = blockIdx.x, tid = threadIdx.x;
    const float* lr = logits + (size_t)row * V;
    float mx = -__builtin_inff();
    for (int i = tid; i < V; i += 256) mx = fmaxf(mx, lr[i]);
    red[tid] = mx; __syncthreads();
    for (int off = 128; off > 0; off >>= 1) {
        if (tid < off) red[tid] = fmaxf(red[tid], red[tid + off]);
        __syncthreads();
    }
    mx = red[0]; __syncthreads();
    float se = 0.f;
    for (int i = tid; i < V; i += 256) se += expf(lr[i] - mx);
    red[tid] = se; __syncthreads();
    for (int off = 128; off > 0; off >>= 1) {
        if (tid < off) red[tid] += red[tid + off];
        __syncthreads();
    }
    if (tid == 0) {
        int tgt = targets[row];
        rowloss[row] = -(lr[tgt] - mx - logf(red[0]));
    }
}

__global__ __launch_bounds__(256) void loss_final_kernel(
    const float* __restrict__ rowloss, float* __restrict__ out, int n)
{
    __shared__ float red[256];
    float s = 0.f;
    for (int i = threadIdx.x; i < n; i += 256) s += rowloss[i];
    red[threadIdx.x] = s; __syncthreads();
    for (int off = 128; off > 0; off >>= 1) {
        if (threadIdx.x < off) red[threadIdx.x] += red[threadIdx.x + off];
        __syncthreads();
    }
    if (threadIdx.x == 0) out[0] = red[0] / (float)n;
}

// ---------------------------------------------------------------------------
extern "C" void kernel_launch(void* const* d_in, const int* in_sizes, int n_in,
                              void* d_out, int out_size, void* d_ws, size_t ws_size,
                              hipStream_t stream)
{
    (void)in_sizes; (void)n_in; (void)out_size; (void)ws_size;
    constexpr int Bb = 4, T = 2048, E = 1024, H = 16, Dh = 64, L = 4, FF = 4096, V = 32000;
    constexpr int BT = Bb * T;   // 8192

    const int*   idx     = (const int*)  d_in[0];
    const int*   targets = (const int*)  d_in[1];
    const float* tok_emb = (const float*)d_in[2];
    const float* pos_emb = (const float*)d_in[3];
    const float* Wq      = (const float*)d_in[4];
    const float* Wk      = (const float*)d_in[5];
    const float* Wv      = (const float*)d_in[6];
    const float* Wo      = (const float*)d_in[7];
    const float* bo      = (const float*)d_in[8];
    const float* ln1_g   = (const float*)d_in[9];
    const float* ln1_b   = (const float*)d_in[10];
    const float* W1      = (const float*)d_in[11];
    const float* b1      = (const float*)d_in[12];
    const float* W2      = (const float*)d_in[13];
    const float* b2      = (const float*)d_in[14];
    const float* ln2_g   = (const float*)d_in[15];
    const float* ln2_b   = (const float*)d_in[16];
    const float* lnf_g   = (const float*)d_in[17];
    const float* lnf_b   = (const float*)d_in[18];
    const float* lm_w    = (const float*)d_in[19];
    const float* lm_b    = (const float*)d_in[20];

    char* ws = (char*)d_ws;
    size_t off = 0;
    auto alloc = [&](size_t bytes) -> void* {
        void* p = ws + off;
        off += (bytes + 255) & ~(size_t)255;
        return p;
    };

    float* x    = (float*)alloc((size_t)BT * E * 4);
    float* res  = (float*)alloc((size_t)BT * E * 4);
    float* qb   = (float*)alloc((size_t)BT * E * 4);
    float* kb   = (float*)alloc((size_t)BT * E * 4);
    float* vb   = (float*)alloc((size_t)BT * E * 4);
    float* attn = (float*)alloc((size_t)BT * E * 4);
    _Float16* a16 = (_Float16*)alloc((size_t)BT * E * 2);
    _Float16* h16 = (_Float16*)alloc((size_t)BT * FF * 2);
    _Float16* wqP = (_Float16*)alloc((size_t)L * Dh * Dh * 2);
    _Float16* wkP = (_Float16*)alloc((size_t)L * Dh * Dh * 2);
    _Float16* wvP = (_Float16*)alloc((size_t)L * Dh * Dh * 2);
    _Float16* woP = (_Float16*)alloc((size_t)L * E * E * 2);
    _Float16* w1P = (_Float16*)alloc((size_t)L * FF * E * 2);
    _Float16* w2P = (_Float16*)alloc((size_t)L * E * FF * 2);
    _Float16* lmP = (_Float16*)alloc((size_t)V * E * 2);
    float* rowloss = (float*)alloc((size_t)BT * 4);

    const dim3 blk(256);

    // --- pack weights to f16 WMMA-fragment order (once per launch) ---
    pack_w_kernel<<<dim3((Dh * Dh + 255) / 256, L), blk, 0, stream>>>(Wq, wqP, Dh, Dh);
    pack_w_kernel<<<dim3((Dh * Dh + 255) / 256, L), blk, 0, stream>>>(Wk, wkP, Dh, Dh);
    pack_w_kernel<<<dim3((Dh * Dh + 255) / 256, L), blk, 0, stream>>>(Wv, wvP, Dh, Dh);
    pack_w_kernel<<<dim3((E * E + 255) / 256,  L), blk, 0, stream>>>(Wo, woP, E, E);
    pack_w_kernel<<<dim3((FF * E + 255) / 256, L), blk, 0, stream>>>(W1, w1P, FF, E);
    pack_w_kernel<<<dim3((E * FF + 255) / 256, L), blk, 0, stream>>>(W2, w2P, E, FF);
    pack_w_kernel<<<dim3((V * E + 255) / 256,  1), blk, 0, stream>>>(lm_w, lmP, V, E);

    // --- embedding ---
    embed_kernel<<<BT, blk, 0, stream>>>(idx, tok_emb, pos_emb, x, T);

    const int nXE = BT * E;
    for (int l = 0; l < L; ++l) {
        // QKV: x viewed as [BT*H, 64] rows, shared 64x64 head weights (NT=2)
        cvt_f16_kernel<<<(nXE + 255) / 256, blk, 0, stream>>>(x, a16, nXE);
        dim3 gq(BT * H / BM, Dh / 64);   // (1024, 1)
        gemm_f16_wmma<2><<<gq, blk, 0, stream>>>(a16, wqP + (size_t)l * Dh * Dh, qb, nullptr, nullptr, BT * H, Dh, Dh, 0);
        gemm_f16_wmma<2><<<gq, blk, 0, stream>>>(a16, wkP + (size_t)l * Dh * Dh, kb, nullptr, nullptr, BT * H, Dh, Dh, 0);
        gemm_f16_wmma<2><<<gq, blk, 0, stream>>>(a16, wvP + (size_t)l * Dh * Dh, vb, nullptr, nullptr, BT * H, Dh, Dh, 0);

        attention_kernel<<<dim3(T / 8, Bb * H), blk, 0, stream>>>(qb, kb, vb, attn, T, H);

        // out projection + residual-LN (NT=4 -> 128-wide N tiles)
        cvt_f16_kernel<<<(nXE + 255) / 256, blk, 0, stream>>>(attn, a16, nXE);
        gemm_f16_wmma<4><<<dim3(BT / BM, E / 128), blk, 0, stream>>>(a16, woP + (size_t)l * E * E, res, nullptr, bo + (size_t)l * E, BT, E, E, 0);
        ln_kernel<<<BT, blk, 0, stream>>>(res, x, ln1_g + (size_t)l * E, ln1_b + (size_t)l * E, x);

        // FFN: GELU fused into GEMM1 epilogue, f16 hidden feeds GEMM2 directly
        cvt_f16_kernel<<<(nXE + 255) / 256, blk, 0, stream>>>(x, a16, nXE);
        gemm_f16_wmma<4><<<dim3(BT / BM, FF / 128), blk, 0, stream>>>(a16, w1P + (size_t)l * FF * E, nullptr, h16, b1 + (size_t)l * FF, BT, FF, E, 1);
        gemm_f16_wmma<4><<<dim3(BT / BM, E / 128), blk, 0, stream>>>(h16, w2P + (size_t)l * E * FF, res, nullptr, b2 + (size_t)l * E, BT, E, FF, 0);
        ln_kernel<<<BT, blk, 0, stream>>>(res, x, ln2_g + (size_t)l * E, ln2_b + (size_t)l * E, x);
    }

    // --- final LN + LM head (M=8192, N=32000, K=1024) ---
    ln_kernel<<<BT, blk, 0, stream>>>(x, nullptr, lnf_g, lnf_b, res);
    cvt_f16_kernel<<<(nXE + 255) / 256, blk, 0, stream>>>(res, a16, nXE);
    float* logits = (float*)d_out;
    gemm_f16_wmma<4><<<dim3(BT / BM, V / 128), blk, 0, stream>>>(a16, lmP, logits, nullptr, lm_b, BT, V, E, 0);

    // --- loss ---
    loss_row_kernel<<<BT, blk, 0, stream>>>(logits, targets, rowloss, V);
    loss_final_kernel<<<1, blk, 0, stream>>>(rowloss, logits + (size_t)BT * V, BT);
}
